// IBT_semseg_33028298506413
// MI455X (gfx1250) — compile-verified, matrix operand order checked by reading, not run on hardware
//
#include <hip/hip_runtime.h>
#include <math.h>

typedef __attribute__((ext_vector_type(16))) _Float16 v16h;
typedef __attribute__((ext_vector_type(8)))  _Float16 v8h;
typedef __attribute__((ext_vector_type(8)))  float    v8f;

#define K_NEIGH 20
#define NPTS    4096
#define BATCH   4
#define M1      (BATCH * NPTS)            /* 16384  */
#define M2      (BATCH * NPTS * K_NEIGH)  /* 327680 */

// ---------------------------------------------------------------------------
// Stage one (32K x 32N-pair) B slice into registers as f16 (WMMA lane layout).
// tail=false: branch/select-free (addresses pre-clamped by caller).
// ---------------------------------------------------------------------------
__device__ __forceinline__ void stage_b(const float* __restrict__ B, int Bld,
                                        int kt, int K, int koff,
                                        int cn0c, int cn1c, bool tail,
                                        v8h& p0, v8h& p1)
{
  if (!tail) {
    const float* Bp = B + (kt + koff) * Bld;
#pragma unroll
    for (int j = 0; j < 8; ++j) {
      p0[j] = (_Float16)Bp[j * Bld + cn0c];
      p1[j] = (_Float16)Bp[j * Bld + cn1c];
    }
  } else {
#pragma unroll
    for (int j = 0; j < 8; ++j) {
      int kk = kt + koff + j;
      int kc = kk < K ? kk : 0;
      float v0 = B[kc * Bld + cn0c];
      float v1 = B[kc * Bld + cn1c];
      p0[j] = (_Float16)(kk < K ? v0 : 0.f);
      p1[j] = (_Float16)(kk < K ? v1 : 0.f);
    }
  }
}

__device__ __forceinline__ v16h load_a(const float* __restrict__ Arow,
                                       int kt, int K, int hb, bool tail)
{
  v16h a;
#pragma unroll
  for (int e = 0; e < 16; ++e) {
    int kk = kt + ((e < 8) ? (hb + e) : (16 + hb + (e - 8)));
    if (!tail) {
      a[e] = (_Float16)Arow[kk];
    } else {
      int kc = kk < K ? kk : 0;
      float av = Arow[kc];
      a[e] = (_Float16)(kk < K ? av : 0.f);
    }
  }
  return a;
}

// ---------------------------------------------------------------------------
// WMMA GEMM:  D[M x N] = A x B (+ bias[row]); row-major, unit column stride.
// Block = 128 threads = 4 waves; macro-tile 64(M) x 64(N); 4 WMMAs per wave
// per 32-K step. B panel double-buffered in LDS (pre-swizzled to the WMMA
// lane layout): next panel's global loads overlap current step's WMMAs and
// only ONE workgroup barrier is needed per k-step. Out-of-range rows/cols
// are address-clamped (they only feed unstored lanes); K-tail zero-fills via
// a wave-uniform flag.
// ---------------------------------------------------------------------------
__global__ __launch_bounds__(128)
void gemm_wmma(const float* __restrict__ A, const float* __restrict__ B,
               float* __restrict__ D, int M, int N, int K,
               int Ald, int Bld, int Dld, const float* __restrict__ bias)
{
  __shared__ __align__(32) _Float16 lb[4096];  // 2 x (4 x 32Kx16N subtile) buffers
  int bn0 = blockIdx.x * 64;
  int bm0 = blockIdx.y * 64;
  int t = threadIdx.x;
  int w = t >> 5;
  int lane = t & 31;
  int l16 = lane & 15;
  int hb = (lane >> 4) * 8;                 // K sub-block base for this lane-half
  int arow = bm0 + w * 16 + l16;
  int arowc = arow < M ? arow : (M - 1);    // clamp: bad rows never stored
  const float* Arow = A + arowc * Ald;

  // cooperative B loader: thread feeds subtiles s0 and s0+2, consumer lane bl,
  // elements e0..e0+7 (contiguous k-run).
  int s0 = t >> 6;
  int r  = t & 63;
  int bl = r >> 1;
  int e0 = (r & 1) * 8;
  int bg = bl >> 4;
  int koff = bg * 8 + (e0 ? 16 : 0);
  int cn0 = bn0 + (bl & 15) + s0 * 16;
  int cn1 = cn0 + 32;
  int cn0c = cn0 < N ? cn0 : (N - 1);       // clamp: bad cols never stored
  int cn1c = cn1 < N ? cn1 : (N - 1);
  _Float16* st0 = &lb[s0 * 512 + bl * 16 + e0];
  _Float16* st1 = st0 + 1024;

  v8f acc0 = {}, acc1 = {}, acc2 = {}, acc3 = {};
  int nsteps = (K + 31) >> 5;

  // prologue: stage step 0 into buffer 0
  {
    v8h p0, p1;
    stage_b(B, Bld, 0, K, koff, cn0c, cn1c, K < 32, p0, p1);
    *(v8h*)st0 = p0;
    *(v8h*)st1 = p1;
  }
  __syncthreads();

  for (int s = 0; s < nsteps; ++s) {
    int kt = s << 5;
    int cur = s & 1;
    bool last = (s + 1) == nsteps;
    v8h q0, q1;
    if (!last)                               // prefetch next panel (overlaps WMMAs)
      stage_b(B, Bld, kt + 32, K, koff, cn0c, cn1c, (kt + 64) > K, q0, q1);
    v16h a = load_a(Arow, kt, K, hb, (kt + 32) > K);
    const _Float16* lc = &lb[cur * 2048];
    v16h b0 = *(const v16h*)&lc[lane * 16];
    v16h b1 = *(const v16h*)&lc[512 + lane * 16];
    v16h b2 = *(const v16h*)&lc[1024 + lane * 16];
    v16h b3 = *(const v16h*)&lc[1536 + lane * 16];
    acc0 = __builtin_amdgcn_wmma_f32_16x16x32_f16(false, a, false, b0, (short)0, acc0, false, false);
    acc1 = __builtin_amdgcn_wmma_f32_16x16x32_f16(false, a, false, b1, (short)0, acc1, false, false);
    acc2 = __builtin_amdgcn_wmma_f32_16x16x32_f16(false, a, false, b2, (short)0, acc2, false, false);
    acc3 = __builtin_amdgcn_wmma_f32_16x16x32_f16(false, a, false, b3, (short)0, acc3, false, false);
    if (!last) {
      _Float16* t0 = st0 + (1 - cur) * 2048;
      _Float16* t1 = st1 + (1 - cur) * 2048;
      *(v8h*)t0 = q0;
      *(v8h*)t1 = q1;
      __syncthreads();                       // one barrier per k-step
    }
  }

  int rowb = bm0 + w * 16 + hb;
  int c0 = bn0 + l16;
#pragma unroll
  for (int r8 = 0; r8 < 8; ++r8) {
    int row = rowb + r8;
    if (row < M) {
      float bb = bias ? bias[row] : 0.f;
      if (c0      < N) D[row * Dld + c0     ] = acc0[r8] + bb;
      if (c0 + 16 < N) D[row * Dld + c0 + 16] = acc1[r8] + bb;
      if (c0 + 32 < N) D[row * Dld + c0 + 32] = acc2[r8] + bb;
      if (c0 + 48 < N) D[row * Dld + c0 + 48] = acc3[r8] + bb;
    }
  }
}

// ---------------------------------------------------------------------------
// kNN top-20 (largest pd = -||a-b||^2) on a 3-channel slice of x [B,9,N].
// ---------------------------------------------------------------------------
__global__ void knn_topk(const float* __restrict__ X, int c0,
                         int* __restrict__ oidx, float* __restrict__ odis)
{
  int pid = blockIdx.x;
  int b = pid / NPTS, n = pid % NPTS;
  int lane = threadIdx.x;
  const float* base = X + (long)b * 9 * NPTS + (long)c0 * NPTS;
  float px = base[n], py = base[NPTS + n], pz = base[2 * NPTS + n];
  float bd[K_NEIGH]; int bi[K_NEIGH];
#pragma unroll
  for (int s = 0; s < K_NEIGH; ++s) { bd[s] = -3.4e38f; bi[s] = 0; }
  for (int m = lane; m < NPTS; m += 32) {
    float dx = px - base[m], dy = py - base[NPTS + m], dz = pz - base[2 * NPTS + m];
    float pd = -(dx * dx + dy * dy + dz * dz);
    if (pd > bd[K_NEIGH - 1]) {
      float v = pd; int vi = m;
#pragma unroll
      for (int s2 = 0; s2 < K_NEIGH; ++s2) {
        if (v > bd[s2]) { float tv = bd[s2]; int ti = bi[s2]; bd[s2] = v; bi[s2] = vi; v = tv; vi = ti; }
      }
    }
  }
  __shared__ float sd[32 * K_NEIGH];
  __shared__ int   si[32 * K_NEIGH];
  __shared__ int   scur[32];
#pragma unroll
  for (int s2 = 0; s2 < K_NEIGH; ++s2) { sd[lane * K_NEIGH + s2] = bd[s2]; si[lane * K_NEIGH + s2] = bi[s2]; }
  scur[lane] = 0;
  __syncthreads();
  if (lane == 0) {
    long ob = (long)pid * K_NEIGH;
    for (int r = 0; r < K_NEIGH; ++r) {
      float best = -3.4e38f; int bsl = 0;
      for (int l = 0; l < 32; ++l) {
        float v = (scur[l] < K_NEIGH) ? sd[l * K_NEIGH + scur[l]] : -3.4e38f;
        if (v > best) { best = v; bsl = l; }
      }
      oidx[ob + r] = si[bsl * K_NEIGH + scur[bsl]];
      odis[ob + r] = best;
      scur[bsl]++;
    }
  }
}

// ---------------------------------------------------------------------------
// Graph features
// ---------------------------------------------------------------------------
__global__ void graph_feat0(const float* __restrict__ Xin, const int* __restrict__ idx,
                            float* __restrict__ x0g)
{
  long i = (long)blockIdx.x * blockDim.x + threadIdx.x;
  if (i >= (long)M2) return;
  int b = (int)(i / (NPTS * K_NEIGH));
  int r = (int)(i % (NPTS * K_NEIGH));
  int n = r / K_NEIGH;
  int j = idx[i];
  const float* xb = Xin + (long)b * 9 * NPTS;
#pragma unroll
  for (int c = 0; c < 9; ++c) {
    float d = xb[c * NPTS + j] - xb[c * NPTS + n];   // neighbor - center
    x0g[(long)c * M2 + i]       = d;
    x0g[(long)(9 + c) * M2 + i] = d;
  }
}

__global__ void graph_feat(const float* __restrict__ Xin, const float* __restrict__ feat,
                           const int* __restrict__ idx, const float* __restrict__ dis,
                           float* __restrict__ pos, float* __restrict__ xf)
{
  long i = (long)blockIdx.x * blockDim.x + threadIdx.x;
  if (i >= (long)M2) return;
  int b = (int)(i / (NPTS * K_NEIGH));
  int r = (int)(i % (NPTS * K_NEIGH));
  int n = r / K_NEIGH;
  int j = idx[i];
  const float* xb = Xin + (long)b * 9 * NPTS;
#pragma unroll
  for (int c = 0; c < 3; ++c)
    pos[(long)c * M2 + i] = xb[c * NPTS + n] - xb[c * NPTS + j];  // center - neighbor
  pos[3L * M2 + i] = dis[i];
  long mc = (long)b * NPTS + n, mj = (long)b * NPTS + j;
  for (int c = 0; c < 128; ++c) {
    float d = feat[(long)c * M1 + mj] - feat[(long)c * M1 + mc];
    pos[(long)(4 + c) * M2 + i] = d;
    xf[(long)c * M2 + i]        = d;
  }
}

// ---------------------------------------------------------------------------
// BatchNorm stats + apply
// ---------------------------------------------------------------------------
__global__ void rowstats(const float* __restrict__ x, float* __restrict__ mean,
                         float* __restrict__ rstd, long M)
{
  int c = blockIdx.x, tid = threadIdx.x;
  __shared__ float ss[256], sq[256];
  float s = 0.f, q = 0.f;
  for (long m = tid; m < M; m += 256) { float v = x[(long)c * M + m]; s += v; q += v * v; }
  ss[tid] = s; sq[tid] = q; __syncthreads();
  for (int o = 128; o > 0; o >>= 1) {
    if (tid < o) { ss[tid] += ss[tid + o]; sq[tid] += sq[tid + o]; }
    __syncthreads();
  }
  if (tid == 0) {
    float m = ss[0] / (float)M;
    float v = sq[0] / (float)M - m * m;
    if (v < 0.f) v = 0.f;
    mean[c] = m; rstd[c] = rsqrtf(v + 1e-5f);
  }
}

__global__ void bn_act(const float* __restrict__ x, float* __restrict__ out,
                       const float* __restrict__ mean, const float* __restrict__ rstd,
                       const float* __restrict__ g, const float* __restrict__ bta,
                       long M, int act, const float* __restrict__ res1,
                       const float* __restrict__ res2, long total)
{
  long i = (long)blockIdx.x * blockDim.x + threadIdx.x;
  if (i >= total) return;
  int c = (int)(i / M);
  float y = g[c] * (x[i] - mean[c]) * rstd[c] + bta[c];
  if (act == 1) y = (y > 0.f) ? y : 0.2f * y;
  else if (act == 2) y = fmaxf(y, 0.f);
  if (res1) y += res1[i];
  if (res2) y += res2[i];
  out[i] = y;
}

// ---------------------------------------------------------------------------
// Reductions / aggregation
// ---------------------------------------------------------------------------
__global__ void max_over_k(const float* __restrict__ src, float* __restrict__ dst, int C)
{
  long i = (long)blockIdx.x * blockDim.x + threadIdx.x;
  if (i >= (long)C * M1) return;
  int c = (int)(i / M1), m = (int)(i % M1);
  const float* p = src + (long)c * M2 + (long)m * K_NEIGH;
  float mx = p[0];
#pragma unroll
  for (int k = 1; k < K_NEIGH; ++k) mx = fmaxf(mx, p[k]);
  dst[(long)c * M1 + m] = mx;
}

__global__ void softmax_k_agg(const float* __restrict__ E, const float* __restrict__ XK,
                              float* __restrict__ out)
{
  long i = (long)blockIdx.x * blockDim.x + threadIdx.x;
  if (i >= 128L * M1) return;
  int c = (int)(i / M1), m = (int)(i % M1);
  const float* e = E  + (long)c * M2 + (long)m * K_NEIGH;
  const float* x = XK + (long)c * M2 + (long)m * K_NEIGH;
  float mx = e[0];
#pragma unroll
  for (int k = 1; k < K_NEIGH; ++k) mx = fmaxf(mx, e[k]);
  float den = 0.f, num = 0.f;
#pragma unroll
  for (int k = 0; k < K_NEIGH; ++k) {
    float s = __expf(e[k] - mx);
    den += s; num += s * x[k];
  }
  out[(long)c * M1 + m] = num / den;
}

__global__ void max_over_n(const float* __restrict__ src, float* __restrict__ dstBroad,
                           float* __restrict__ dstSmall, int mode)
{
  int c = blockIdx.x / BATCH, b = blockIdx.x % BATCH;
  int tid = threadIdx.x;
  const float* p = src + (long)c * M1 + (long)b * NPTS;
  __shared__ float red[256];
  float mx = -3.4e38f;
  for (int n = tid; n < NPTS; n += 256) mx = fmaxf(mx, p[n]);
  red[tid] = mx; __syncthreads();
  for (int o = 128; o > 0; o >>= 1) {
    if (tid < o) red[tid] = fmaxf(red[tid], red[tid + o]);
    __syncthreads();
  }
  mx = red[0];
  if (mode == 0) { if (tid == 0) dstSmall[c * BATCH + b] = mx; }
  else {
    float* q = dstBroad + (long)c * M1 + (long)b * NPTS;
    for (int n = tid; n < NPTS; n += 256) q[n] = mx;
  }
}

// ---------------------------------------------------------------------------
// Attention helpers
// ---------------------------------------------------------------------------
__global__ void row_softmax(float* __restrict__ E, int N)
{
  int row = blockIdx.x, tid = threadIdx.x;
  float* r = E + (long)row * N;
  __shared__ float red[256];
  float mx = -3.4e38f;
  for (int i = tid; i < N; i += 256) mx = fmaxf(mx, r[i]);
  red[tid] = mx; __syncthreads();
  for (int o = 128; o > 0; o >>= 1) { if (tid < o) red[tid] = fmaxf(red[tid], red[tid + o]); __syncthreads(); }
  mx = red[0]; __syncthreads();
  float s = 0.f;
  for (int i = tid; i < N; i += 256) { float e = __expf(r[i] - mx); r[i] = e; s += e; }
  red[tid] = s; __syncthreads();
  for (int o = 128; o > 0; o >>= 1) { if (tid < o) red[tid] += red[tid + o]; __syncthreads(); }
  float inv = 1.f / red[0];
  for (int i = tid; i < N; i += 256) r[i] *= inv;
}

__global__ void col_sum(const float* __restrict__ E, float* __restrict__ cs, int N)
{
  int col = blockIdx.x * blockDim.x + threadIdx.x;
  if (col >= N) return;
  float s = 0.f;
  for (int r = 0; r < N; ++r) s += E[(long)r * N + col];
  cs[col] = s;
}

// ---------------------------------------------------------------------------
// Elementwise helpers
// ---------------------------------------------------------------------------
__global__ void add2(const float* __restrict__ a, const float* __restrict__ b,
                     float* __restrict__ o, long n)
{ long i = (long)blockIdx.x * blockDim.x + threadIdx.x; if (i < n) o[i] = a[i] + b[i]; }

__global__ void subf(const float* __restrict__ a, float* __restrict__ o, long n)
{ long i = (long)blockIdx.x * blockDim.x + threadIdx.x; if (i < n) o[i] = a[i] - o[i]; }

__global__ void vprep(float* __restrict__ v, const float* __restrict__ pos,
                      const float* __restrict__ edge, long n)
{
  long i = (long)blockIdx.x * blockDim.x + threadIdx.x;
  if (i < n) v[i] = (v[i] + pos[i]) * (1.f / (1.f + __expf(-edge[i])));
}

__global__ void colscale(float* __restrict__ xr, const float* __restrict__ cs)
{
  long i = (long)blockIdx.x * blockDim.x + threadIdx.x;
  if (i >= 128L * NPTS) return;
  int c = (int)(i / NPTS), m = (int)(i % NPTS);
  xr[(long)c * M1 + m] *= 1.f / (1e-9f + cs[m]);
}

__global__ void transpose_in(const float* __restrict__ X, float* __restrict__ X9)
{
  long i = (long)blockIdx.x * blockDim.x + threadIdx.x;
  if (i >= 9L * M1) return;
  int c = (int)(i / M1), m = (int)(i % M1);
  int b = m / NPTS, n = m % NPTS;
  X9[i] = X[((long)b * 9 + c) * NPTS + n];
}

// Q [32, M1] -> QT [M1, 32] row-major (energy GEMM gets unit-stride K)
__global__ void transQ(const float* __restrict__ Q, float* __restrict__ QT)
{
  long i = (long)blockIdx.x * blockDim.x + threadIdx.x;
  if (i >= 32L * M1) return;
  int c = (int)(i / M1), m = (int)(i % M1);
  QT[(long)m * 32 + c] = Q[i];
}

__global__ void transform_apply(const float* __restrict__ X9, const float* __restrict__ T9,
                                float* __restrict__ XT)
{
  long i = (long)blockIdx.x * blockDim.x + threadIdx.x;
  if (i >= 9L * M1) return;
  int d = (int)(i / M1), m = (int)(i % M1);
  int b = m / NPTS;
  float s = 0.f;
#pragma unroll
  for (int c = 0; c < 9; ++c) s += X9[(long)c * M1 + m] * T9[(c * 9 + d) * BATCH + b];
  XT[i] = s;
}

__global__ void reorder_out(const float* __restrict__ O13, float* __restrict__ out)
{
  long i = (long)blockIdx.x * blockDim.x + threadIdx.x;
  if (i >= 13L * M1) return;
  int c = (int)(i / M1), m = (int)(i % M1);
  int b = m / NPTS, n = m % NPTS;
  out[((long)b * 13 + c) * NPTS + n] = O13[i];
}

// ---------------------------------------------------------------------------
// Host orchestration
// ---------------------------------------------------------------------------
extern "C" void kernel_launch(void* const* d_in, const int* in_sizes, int n_in,
                              void* d_out, int out_size, void* d_ws, size_t ws_size,
                              hipStream_t stream)
{
  (void)in_sizes; (void)n_in; (void)out_size; (void)ws_size;
  auto PR = [&](int i) { return (const float*)d_in[i]; };
  const float* Xin = PR(99);                     // x [B,9,N]

  float* W = (float*)d_ws;
  long off = 0;
  auto alloc = [&](long n) { float* p = W + off; off += (n + 63) & ~63L; return p; };
  float* X9   = alloc(9L * M1);
  float* XT   = alloc(9L * M1);
  float* x_p  = alloc(128L * M1);
  float* x0b  = alloc(128L * M1);
  float* x1b  = alloc(128L * M1);
  float* x2b  = alloc(128L * M1);
  float* POSb = alloc(128L * M1);
  float* XPb  = alloc(128L * M1);
  float* Vb   = alloc(128L * M1);
  float* XR   = alloc(128L * M1);
  float* TT   = alloc(128L * M1);
  float* Qb   = alloc(32L * M1);
  float* QTb  = alloc(32L * M1);
  float* Kb   = alloc(32L * M1);
  float* AG   = alloc(256L * M1);
  int*   IDX0 = (int*)alloc(M2);
  float* DIS0 = alloc(M2);
  int*   IDXa = (int*)alloc(M2);
  float* DISa = alloc(M2);
  float* TB   = alloc(1024L * BATCH);
  float* SM1  = alloc(512L * BATCH);
  float* SM2  = alloc(256L * BATCH);
  float* SM3  = alloc(128L * BATCH);
  float* CS   = alloc(NPTS);
  float* MEANb = alloc(2048);
  float* RSTDb = alloc(2048);
  float* POSA = alloc(132L * M2);   // also: T1, T3, XK, C7
  float* CC   = alloc(192L * M2);   // also: X0g, TA, p-chain, ENERGY, H1, C8..out
  float* EAR  = alloc(128L * M2);   // also: T2, Y64, E, H2
  float* ENERGY = CC;

  auto ew = [&](long n) { return dim3((unsigned)((n + 255) / 256)); };
  auto gemm = [&](const float* A, const float* Bm, float* Dm, int M, int Nn, int K,
                  int Ald, int Bld, int Dld, const float* bias) {
    dim3 grid((unsigned)((Nn + 63) / 64), (unsigned)((M + 63) / 64));
    gemm_wmma<<<grid, 128, 0, stream>>>(A, Bm, Dm, M, Nn, K, Ald, Bld, Dld, bias);
  };
  auto wgemm = [&](const float* Wt, const float* Xm, float* Dm, int O, long Mc, int K,
                   int ldB, const float* bias) {
    gemm(Wt, Xm, Dm, O, (int)Mc, K, K, ldB, ldB, bias);
  };
  auto stats_bn = [&](const float* x, float* out, int C, long M, int act,
                      const float* g, const float* bta, const float* r1, const float* r2) {
    rowstats<<<dim3(C), 256, 0, stream>>>(x, MEANb, RSTDb, M);
    long tot = (long)C * M;
    bn_act<<<ew(tot), 256, 0, stream>>>(x, out, MEANb, RSTDb, g, bta, M, act, r1, r2, tot);
  };

  transpose_in<<<ew(9L * M1), 256, 0, stream>>>(Xin, X9);
  knn_topk<<<dim3(M1), 32, 0, stream>>>(Xin, 6, IDX0, DIS0);
  graph_feat0<<<ew((long)M2), 256, 0, stream>>>(Xin, IDX0, CC);   // X0g [18,M2]

  // transform_net
  wgemm(PR(92), CC,  POSA, 64,   M2, 18,  M2, nullptr);
  stats_bn(POSA, POSA, 64, M2, 1, PR(82), PR(83), nullptr, nullptr);
  wgemm(PR(93), POSA, EAR, 128,  M2, 64,  M2, nullptr);
  stats_bn(EAR, EAR, 128, M2, 1, PR(84), PR(85), nullptr, nullptr);
  max_over_k<<<ew(128L * M1), 256, 0, stream>>>(EAR, CC, 128);
  wgemm(PR(94), CC, POSA, 1024, M1, 128, M1, nullptr);
  stats_bn(POSA, POSA, 1024, M1, 1, PR(86), PR(87), nullptr, nullptr);
  max_over_n<<<dim3(1024 * BATCH), 256, 0, stream>>>(POSA, nullptr, TB, 0);
  wgemm(PR(95), TB, SM1, 512, BATCH, 1024, BATCH, nullptr);
  stats_bn(SM1, SM1, 512, BATCH, 1, PR(88), PR(89), nullptr, nullptr);
  wgemm(PR(96), SM1, SM2, 256, BATCH, 512, BATCH, nullptr);
  stats_bn(SM2, SM2, 256, BATCH, 1, PR(90), PR(91), nullptr, nullptr);
  wgemm(PR(98), SM2, SM3, 81, BATCH, 256, BATCH, PR(97));
  transform_apply<<<ew(9L * M1), 256, 0, stream>>>(X9, SM3, XT);

  // p1..p3 -> x_p
  wgemm(PR(40), XT, CC, 32, M1, 9, M1, nullptr);
  stats_bn(CC, CC, 32, M1, 1, PR(41), PR(42), nullptr, nullptr);
  wgemm(PR(43), CC, CC + 32L * M1, 64, M1, 32, M1, nullptr);
  stats_bn(CC + 32L * M1, CC + 32L * M1, 64, M1, 1, PR(44), PR(45), nullptr, nullptr);
  wgemm(PR(46), CC + 32L * M1, x_p, 128, M1, 64, M1, nullptr);
  stats_bn(x_p, x_p, 128, M1, 1, PR(47), PR(48), nullptr, nullptr);

  knn_topk<<<dim3(M1), 32, 0, stream>>>(Xin, 0, IDXa, DISa);

  auto edge_sa_block = [&](const float* featIn, int mlpW, int mlpG, int convW, int convG,
                           int scoreW, int aggW, int aggG, int s, float* outBuf) {
    float* XF = CC + 64L * M2;
    graph_feat<<<ew((long)M2), 256, 0, stream>>>(Xin, featIn, IDXa, DISa, POSA, XF);
    wgemm(PR(mlpW), POSA, EAR, 64, M2, 132, M2, nullptr);
    stats_bn(EAR, CC, 64, M2, 1, PR(mlpG), PR(mlpG + 1), nullptr, nullptr);
    wgemm(PR(convW), CC, POSA, 128, M2, 192, M2, nullptr);
    stats_bn(POSA, POSA, 128, M2, 1, PR(convG), PR(convG + 1), nullptr, nullptr);
    max_over_k<<<ew(128L * M1), 256, 0, stream>>>(POSA, AG, 128);
    wgemm(PR(scoreW), POSA, EAR, 128, M2, 128, M2, nullptr);
    softmax_k_agg<<<ew(128L * M1), 256, 0, stream>>>(EAR, POSA, AG + 128L * M1);
    wgemm(PR(aggW), AG, outBuf, 128, M1, 256, M1, nullptr);
    stats_bn(outBuf, outBuf, 128, M1, 1, PR(aggG), PR(aggG + 1), nullptr, nullptr);
    // ---- sa_layer(x = featIn, t = sigmoid(outBuf)) ----
    gemm(PR(s + 4), X9, POSb, 128, M1, 3, 3, M1, M1, PR(s + 3));
    add2<<<ew(128L * M1), 256, 0, stream>>>(featIn, POSb, XPb, 128L * M1);
    wgemm(PR(s + 5), XPb, Qb, 32, M1, 128, M1, nullptr);
    transQ<<<ew(32L * M1), 256, 0, stream>>>(Qb, QTb);
    wgemm(PR(s + 2), XPb, Kb, 32, M1, 128, M1, nullptr);
    wgemm(PR(s + 9), XPb, Vb, 128, M1, 128, M1, PR(s + 8));
    vprep<<<ew(128L * M1), 256, 0, stream>>>(Vb, POSb, outBuf, 128L * M1);
    for (int b = 0; b < BATCH; ++b) {
      gemm(QTb + (long)b * NPTS * 32, Kb + (long)b * NPTS, ENERGY,
           NPTS, NPTS, 32, 32, M1, NPTS, nullptr);
      row_softmax<<<dim3(NPTS), 256, 0, stream>>>(ENERGY, NPTS);
      col_sum<<<dim3(NPTS / 256), 256, 0, stream>>>(ENERGY, CS, NPTS);
      gemm(Vb + (long)b * NPTS, ENERGY, XR + (long)b * NPTS,
           128, NPTS, NPTS, M1, NPTS, M1, nullptr);
      colscale<<<ew(128L * NPTS), 256, 0, stream>>>(XR + (long)b * NPTS, CS);
    }
    subf<<<ew(128L * M1), 256, 0, stream>>>(XPb, XR, 128L * M1);
    wgemm(PR(s + 7), XR, TT, 128, M1, 128, M1, PR(s + 6));
    stats_bn(TT, outBuf, 128, M1, 2, PR(s + 0), PR(s + 1), outBuf, XPb);
  };

  edge_sa_block(x_p, 31, 32, 13, 14, 79, 22, 23, 49, x0b);  // mlp0/conv0/score0/conv4/sa0
  edge_sa_block(x0b, 34, 35, 16, 17, 80, 25, 26, 69, x1b);  // mlp1/conv1/score1/conv5/sa2
  edge_sa_block(x1b, 37, 38, 19, 20, 81, 28, 29, 59, x2b);  // mlp2/conv3/score2/conv6/sa1

  // head
  const long ROW = 128L * M1 * sizeof(float);
  float* H1 = CC;
  max_over_n<<<dim3(128 * BATCH), 256, 0, stream>>>(x_p, H1, nullptr, 1);
  hipMemcpyAsync(H1 + 128L * M1, x0b, ROW, hipMemcpyDeviceToDevice, stream);
  hipMemcpyAsync(H1 + 256L * M1, x1b, ROW, hipMemcpyDeviceToDevice, stream);
  hipMemcpyAsync(H1 + 384L * M1, x2b, ROW, hipMemcpyDeviceToDevice, stream);
  float* C7 = POSA;
  wgemm(PR(4), H1, C7, 1024, M1, 512, M1, nullptr);
  stats_bn(C7, C7, 1024, M1, 1, PR(5), PR(6), nullptr, nullptr);
  float* H2 = EAR;
  max_over_n<<<dim3(1024 * BATCH), 256, 0, stream>>>(C7, H2, nullptr, 1);
  hipMemcpyAsync(H2 + 1024L * M1, x0b, ROW, hipMemcpyDeviceToDevice, stream);
  hipMemcpyAsync(H2 + 1152L * M1, x1b, ROW, hipMemcpyDeviceToDevice, stream);
  hipMemcpyAsync(H2 + 1280L * M1, x2b, ROW, hipMemcpyDeviceToDevice, stream);
  float* C8 = CC + 512L * M1;
  wgemm(PR(7), H2, C8, 512, M1, 1408, M1, nullptr);
  stats_bn(C8, C8, 512, M1, 1, PR(8), PR(9), nullptr, nullptr);
  float* C9 = C8 + 512L * M1;
  wgemm(PR(10), C8, C9, 256, M1, 512, M1, nullptr);
  stats_bn(C9, C9, 256, M1, 1, PR(11), PR(12), nullptr, nullptr);
  float* C10 = C9 + 256L * M1;
  wgemm(PR(0), C9, C10, 128, M1, 256, M1, nullptr);
  stats_bn(C10, C10, 128, M1, 1, PR(1), PR(2), nullptr, nullptr);
  float* O13 = C10 + 128L * M1;
  wgemm(PR(3), C10, O13, 13, M1, 128, M1, nullptr);
  reorder_out<<<ew(13L * M1), 256, 0, stream>>>(O13, (float*)d_out);
}
